// MultiHeadAttention_343597384205
// MI455X (gfx1250) — matrix operand order repaired and checked by reading, not compiled, not run on hardware
//
#include <hip/hip_runtime.h>
#include <hip/hip_bf16.h>

// ---------------------------------------------------------------------------
// MultiHeadAttention (reference semantics: attention across the 32 HEADS per
// token, 32x32 energy, scale 1/sqrt(d_model)=1/64).
//
// Precision: bf16 WMMA (v_wmma_f32_16x16x32_bf16), f32 accumulate.
// Roofline: out-projection GEMM (8192x4096x4096 = 274 GFLOP) dominates;
// ~450 MB HBM traffic (~20us @ 23.3 TB/s) => balanced at bf16 matrix rates.
// Data movement: double-buffered LDS tiles fed by GLOBAL_LOAD_ASYNC_TO_LDS
// (ASYNCcnt) for activations and the Tensor Data Mover (TENSORcnt) for the
// weight tiles.
// ---------------------------------------------------------------------------

typedef __bf16 bf16;
typedef __attribute__((ext_vector_type(16))) __bf16 bf16x16;
typedef __attribute__((ext_vector_type(8)))  __bf16 bf16x8;
typedef __attribute__((ext_vector_type(8)))  float  f32x8;
typedef __attribute__((ext_vector_type(4)))  unsigned int u32x4;
typedef __attribute__((ext_vector_type(8)))  unsigned int u32x8;

#define DM    4096
#define HEADS 32
#define HD    128
#define BATCH 2
#define SEQ   4096
#define NTOK  (BATCH * SEQ)   // 8192 tokens
#define KSTEPS (DM / 32)      // 128 k-chunks in out-projection

static __device__ inline f32x8 wmma_bf16(bf16x16 a, bf16x16 b, f32x8 c) {
  // (neg_a, A, neg_b, B, c_mod, C, reuse_a, reuse_b)
  return __builtin_amdgcn_wmma_f32_16x16x32_bf16(false, a, false, b, (short)0, c,
                                                 false, false);
}

// A-fragment (16x32) per ISA layout: lane<16 -> row M=l, K = {kc+0..7, kc+16..23};
// lane>=16 -> row M=l, K = {kc+8..15, kc+24..31}.
static __device__ inline bf16x16 frag_a_f32(const float* base, int ld, int mbase,
                                            int kc, int lane) {
  const int half = lane >> 4, l = lane & 15;
  const int row = mbase + l;
  const float* p0 = base + row * ld + kc + (half ? 8 : 0);
  const float* p1 = base + row * ld + kc + (half ? 24 : 16);
  bf16x16 r;
#pragma unroll
  for (int i = 0; i < 8; ++i) { r[i] = (bf16)p0[i]; r[i + 8] = (bf16)p1[i]; }
  return r;
}

static __device__ inline bf16x16 frag_a_bf16(const bf16* base, int ld, int mbase,
                                             int kc, int lane) {
  const int half = lane >> 4, l = lane & 15;
  const int row = mbase + l;
  const bf16x8 lo = *(const bf16x8*)(base + row * ld + kc + (half ? 8 : 0));
  const bf16x8 hi = *(const bf16x8*)(base + row * ld + kc + (half ? 24 : 16));
  return __builtin_shufflevector(lo, hi, 0, 1, 2, 3, 4, 5, 6, 7,
                                 8, 9, 10, 11, 12, 13, 14, 15);
}

// B-fragment (32x16): lane<16 -> column N=l holds K = {kc+0..7, kc+16..23},
// lane>=16 -> K = {kc+8..15, kc+24..31}. Row-major (N x K) weights feed this directly.
static __device__ inline bf16x16 frag_b_bf16(const bf16* base, int ld, int nbase,
                                             int kc, int lane) {
  const int half = lane >> 4, l = lane & 15;
  const int n = nbase + l;
  const bf16* p = base + n * ld + kc;
  const bf16x8 lo = *(const bf16x8*)(p + (half ? 8 : 0));
  const bf16x8 hi = *(const bf16x8*)(p + (half ? 24 : 16));
  return __builtin_shufflevector(lo, hi, 0, 1, 2, 3, 4, 5, 6, 7,
                                 8, 9, 10, 11, 12, 13, 14, 15);
}

// Raw LDS byte offset for async/TDM destinations (flat addr low 32 bits; the
// LDS aperture mapping truncates to addr[31:0], ISA 10.2).
static __device__ inline unsigned int lds_off(const void* p) {
  return (unsigned int)(size_t)p;
}

// GLOBAL_LOAD_ASYNC_TO_LDS_B128: vdst = LDS byte address, vaddr = 64-bit global.
static __device__ inline void async_load_b128(unsigned int lds_addr,
                                              const void* gptr) {
  asm volatile("global_load_async_to_lds_b128 %0, %1, off"
               :: "v"(lds_addr), "v"(gptr)
               : "memory");
}

static __device__ inline void wait_asynccnt0() {
  asm volatile("s_wait_asynccnt 0x0" ::: "memory");
}

// TDM: 2D tile load via TENSOR_LOAD_TO_LDS with D# groups packed per
// CDNA5 ISA (group0 128b, group1 256b; 2D => VADDR2/3 NULL).
static __device__ inline void tdm_load_2d_bf16(unsigned int lds_byte_addr,
                                               unsigned long long gaddr,
                                               unsigned int tensor_d0,
                                               unsigned int tensor_d1,
                                               unsigned int tile_d0,
                                               unsigned int tile_d1,
                                               unsigned long long stride0) {
  u32x4 g0;
  g0[0] = 1u;                                   // count=1 (valid user D#)
  g0[1] = lds_byte_addr;                        // lds_addr
  g0[2] = (unsigned int)gaddr;                  // global_addr[31:0]
  g0[3] = ((unsigned int)(gaddr >> 32) & 0x01FFFFFFu) | (2u << 30); // [56:32]|type=2
  u32x8 g1;
  g1[0] = (1u << 16);                           // wg_mask=0, data_size=1 (2B)
  g1[1] = (tensor_d0 & 0xFFFFu) << 16;          // tensor_dim0[15:0]
  g1[2] = (tensor_d0 >> 16) | ((tensor_d1 & 0xFFFFu) << 16);
  g1[3] = (tensor_d1 >> 16) | (tile_d0 << 16);  // tile_dim0
  g1[4] = tile_d1 & 0xFFFFu;                    // tile_dim1 (tile_dim2=0)
  g1[5] = (unsigned int)stride0;                // tensor_dim0_stride[31:0]
  g1[6] = (unsigned int)(stride0 >> 32) & 0xFFFFu;
  g1[7] = 0u;
  asm volatile("tensor_load_to_lds %0, %1" :: "s"(g0), "s"(g1) : "memory");
}

// ---------------------------------------------------------------------------
// Kernel 0: f32 -> bf16 weight conversion (grid-stride)
// ---------------------------------------------------------------------------
__global__ void mha_cvt_bf16_kernel(const float* __restrict__ src,
                                    bf16* __restrict__ dst, int n) {
  for (int i = blockIdx.x * blockDim.x + threadIdx.x; i < n;
       i += gridDim.x * blockDim.x)
    dst[i] = (bf16)src[i];
}

// ---------------------------------------------------------------------------
// Kernel 1: fused per-token attention. 1 wave == 1 token, 2 waves / block.
// Per-wave LDS: Q(32x128 bf16) | K(32x128) | V^T(128x32) | attn(32x32) = 26624B
// ---------------------------------------------------------------------------
#define WAVE_LDS 26624

__global__ void __launch_bounds__(64)
mha_fused_attn_kernel(const float* __restrict__ q,
                      const float* __restrict__ k,
                      const float* __restrict__ v,
                      const float* __restrict__ mask,
                      const bf16* __restrict__ Wq,
                      const bf16* __restrict__ Wk,
                      const bf16* __restrict__ Wv,
                      bf16* __restrict__ ctx) {
  extern __shared__ char smem[];
  const int lane = threadIdx.x & 31;
  const int wave = threadIdx.x >> 5;
  const int tok  = blockIdx.x * 2 + wave;
  const int half = lane >> 4, l = lane & 15;

  char* wbase = smem + wave * WAVE_LDS;
  bf16* Qb = (bf16*)(wbase);             // 32 x 128, row-major (head, dim)
  bf16* Kb = (bf16*)(wbase + 8192);      // 32 x 128, row-major (head, dim)
  bf16* Vt = (bf16*)(wbase + 16384);     // 128 x 32, row-major (dim, head)
  bf16* Ab = (bf16*)(wbase + 24576);     // 32 x 32 attn probs

  const float* xq = q + (size_t)tok * DM;
  const float* xk = k + (size_t)tok * DM;
  const float* xv = v + (size_t)tok * DM;

  // ---- per-head projections: P(32x128) = X(32x128) @ W^T (shared weights) ----
  // p/mt/kc unrolled (destination select constant-folds); A-fragments hoisted
  // into registers and reused across all 8 N-tiles; nt kept rolled to bound
  // VGPR pressure (no scratch spills).
#pragma unroll
  for (int p = 0; p < 3; ++p) {
    const float* x = (p == 0) ? xq : (p == 1) ? xk : xv;
    const bf16*  W = (p == 0) ? Wq : (p == 1) ? Wk : Wv;
#pragma unroll
    for (int mt = 0; mt < 2; ++mt) {
      bf16x16 afr[4];
#pragma unroll
      for (int kq = 0; kq < 4; ++kq)
        afr[kq] = frag_a_f32(x, HD, mt * 16, kq * 32, lane);
      for (int nt = 0; nt < 8; ++nt) {   // rolled: bounds register pressure
        f32x8 acc = {0.f, 0.f, 0.f, 0.f, 0.f, 0.f, 0.f, 0.f};
#pragma unroll
        for (int kq = 0; kq < 4; ++kq)
          acc = wmma_bf16(afr[kq], frag_b_bf16(W, HD, nt * 16, kq * 32, lane), acc);
        // C layout: VGPR r -> (M = mt*16 + r + 8*half, N = nt*16 + l)
#pragma unroll
        for (int r = 0; r < 8; ++r) {
          const int row = mt * 16 + r + (half ? 8 : 0);  // head
          const int col = nt * 16 + l;                   // dim
          const bf16 val = (bf16)acc[r];
          if (p == 0)       Qb[row * HD + col] = val;
          else if (p == 1)  Kb[row * HD + col] = val;
          else              Vt[col * HEADS + row] = val; // transposed for PV B-frags
        }
      }
    }
  }
  __syncthreads();

  // ---- energy E(32x32) = Q @ K^T / sqrt(d_model) + mask ----
  f32x8 accE[2][2];
#pragma unroll
  for (int mt = 0; mt < 2; ++mt)
#pragma unroll
    for (int nt = 0; nt < 2; ++nt) {
      f32x8 acc = {0.f, 0.f, 0.f, 0.f, 0.f, 0.f, 0.f, 0.f};
#pragma unroll
      for (int kc = 0; kc < HD; kc += 32)
        acc = wmma_bf16(frag_a_bf16(Qb, HD, mt * 16, kc, lane),
                        frag_b_bf16(Kb, HD, nt * 16, kc, lane), acc);
      accE[mt][nt] = acc;
    }

  // ---- softmax over t (32 cols = 2 N-tiles x 16-lane halves) ----
  const float scale = 0.015625f;  // 1/sqrt(4096)
#pragma unroll
  for (int mt = 0; mt < 2; ++mt) {
#pragma unroll
    for (int r = 0; r < 8; ++r) {
      const int row = mt * 16 + r + (half ? 8 : 0);
      float e0 = accE[mt][0][r] * scale + mask[row * HEADS + l];
      float e1 = accE[mt][1][r] * scale + mask[row * HEADS + 16 + l];
      float mx = fmaxf(e0, e1);
#pragma unroll
      for (int off = 8; off >= 1; off >>= 1) mx = fmaxf(mx, __shfl_xor(mx, off));
      const float p0 = __expf(e0 - mx), p1 = __expf(e1 - mx);
      float sm = p0 + p1;
#pragma unroll
      for (int off = 8; off >= 1; off >>= 1) sm += __shfl_xor(sm, off);
      const float inv = 1.0f / sm;
      Ab[row * HEADS + l]      = (bf16)(p0 * inv);
      Ab[row * HEADS + 16 + l] = (bf16)(p1 * inv);
    }
  }
  __syncthreads();

  // ---- O(32x128) = A(32x32) @ V(32x128); K-dim = 32 heads = one chunk ----
  bf16* crow = ctx + (size_t)tok * DM;
  const f32x8 z = {0.f, 0.f, 0.f, 0.f, 0.f, 0.f, 0.f, 0.f};
#pragma unroll
  for (int mt = 0; mt < 2; ++mt) {
    const bf16x16 a = frag_a_bf16(Ab, HEADS, mt * 16, 0, lane);
    for (int nt = 0; nt < 8; ++nt) {     // rolled: bounds register pressure
      f32x8 o = wmma_bf16(a, frag_b_bf16(Vt, HEADS, nt * 16, 0, lane), z);
#pragma unroll
      for (int r = 0; r < 8; ++r) {
        const int row = mt * 16 + r + (half ? 8 : 0);  // head
        const int col = nt * 16 + l;                   // dim
        crow[row * HD + col] = (bf16)o[r];             // (h*128+d) == H*D concat
      }
    }
  }
}

// ---------------------------------------------------------------------------
// Kernel 2: Y = C @ Wo^T + bo.  M=8192, N=4096, K=4096 (274 GFLOP).
// 256-thread WG, tile 256x128, 8 waves of 64x64 (16 wmma / k-step).
// Double-buffered LDS: A tile via GLOBAL_LOAD_ASYNC_TO_LDS_B128 (ASYNCcnt),
// B (weight) tile via the Tensor Data Mover (TENSORcnt), overlapped with WMMA.
// ---------------------------------------------------------------------------
__global__ void __launch_bounds__(256)
mha_out_proj_kernel(const bf16* __restrict__ C,
                    const bf16* __restrict__ Wo,
                    const float* __restrict__ bo,
                    float* __restrict__ out) {
  __shared__ __align__(16) bf16 tileA[2][256 * 32];  // 2 x 16 KB
  __shared__ __align__(16) bf16 tileB[2][128 * 32];  // 2 x  8 KB
  const int tid  = threadIdx.x;
  const int lane = tid & 31;
  const int wave = tid >> 5;
  const int wm = wave >> 1, wn = wave & 1;    // 4 (M) x 2 (N) wave grid
  const int half = lane >> 4, l = lane & 15;
  const int mBase = blockIdx.x * 256;
  const int nBase = blockIdx.y * 128;

  // Stage k-chunk `kc` into buffer `buf`.
  auto stage = [&](int buf, int kc) {
    // A: 256 rows x 32 bf16; one 64B row per thread, 4 async b128 transfers.
    const bf16* srcA = C + (size_t)(mBase + tid) * DM + kc;
    const unsigned int dstA = lds_off(&tileA[buf][tid * 32]);
#pragma unroll
    for (int c = 0; c < 4; ++c)
      async_load_b128(dstA + c * 16, srcA + c * 8);
    // B: 128x32 bf16 tile of Wo via TDM, issued once (wave 0).
    if (wave == 0)
      tdm_load_2d_bf16(lds_off(&tileB[buf][0]),
                       (unsigned long long)(size_t)Wo +
                           2ull * ((size_t)nBase * DM + kc),
                       /*tensor_d0=*/DM, /*tensor_d1=*/DM,
                       /*tile_d0=*/32, /*tile_d1=*/128,
                       /*stride0=*/DM);
  };
  auto wait_stage = [&]() {
    wait_asynccnt0();
    if (wave == 0) __builtin_amdgcn_s_wait_tensorcnt(0);
    __syncthreads();
  };

  f32x8 acc[4][4];
  const f32x8 z = {0.f, 0.f, 0.f, 0.f, 0.f, 0.f, 0.f, 0.f};
#pragma unroll
  for (int i = 0; i < 4; ++i)
#pragma unroll
    for (int j = 0; j < 4; ++j) acc[i][j] = z;

  stage(0, 0);
  wait_stage();

  for (int kt = 0; kt < KSTEPS; ++kt) {
    const int cur = kt & 1;
    if (kt + 1 < KSTEPS) {
      stage(1 - cur, (kt + 1) * 32);
      __builtin_prefetch(C + (size_t)(mBase + tid) * DM + (kt + 2) * 32, 0, 1);
    }

    bf16x16 af[4], bfv[4];
#pragma unroll
    for (int i = 0; i < 4; ++i)
      af[i] = frag_a_bf16(tileA[cur], 32, wm * 64 + i * 16, 0, lane);
#pragma unroll
    for (int j = 0; j < 4; ++j)
      bfv[j] = frag_b_bf16(tileB[cur], 32, wn * 64 + j * 16, 0, lane);
#pragma unroll
    for (int i = 0; i < 4; ++i)
#pragma unroll
      for (int j = 0; j < 4; ++j) acc[i][j] = wmma_bf16(af[i], bfv[j], acc[i][j]);

    wait_stage();  // next buffer resident before it becomes `cur`
  }

  // epilogue: f32 + bias
#pragma unroll
  for (int j = 0; j < 4; ++j) {
    const int col = nBase + wn * 64 + j * 16 + l;
    const float bias = bo[col];
#pragma unroll
    for (int i = 0; i < 4; ++i) {
#pragma unroll
      for (int r = 0; r < 8; ++r) {
        const int row = mBase + wm * 64 + i * 16 + r + (half ? 8 : 0);
        out[(size_t)row * DM + col] = acc[i][j][r] + bias;
      }
    }
  }
}

// ---------------------------------------------------------------------------
// Launch. ws layout: Wq|Wk|Wv bf16 (32KB each), Wo bf16 (32MB), ctx bf16 (64MB)
// ---------------------------------------------------------------------------
extern "C" void kernel_launch(void* const* d_in, const int* in_sizes, int n_in,
                              void* d_out, int out_size, void* d_ws, size_t ws_size,
                              hipStream_t stream) {
  const float* q    = (const float*)d_in[0];
  const float* k    = (const float*)d_in[1];
  const float* v    = (const float*)d_in[2];
  const float* mask = (const float*)d_in[3];
  const float* Wq   = (const float*)d_in[4];
  const float* Wk   = (const float*)d_in[5];
  const float* Wv   = (const float*)d_in[6];
  const float* Wo   = (const float*)d_in[7];
  const float* bo   = (const float*)d_in[8];

  char* ws = (char*)d_ws;
  bf16* Wq_b = (bf16*)(ws);
  bf16* Wk_b = (bf16*)(ws + 32768);
  bf16* Wv_b = (bf16*)(ws + 65536);
  bf16* Wo_b = (bf16*)(ws + 98304);
  bf16* Cb   = (bf16*)(ws + 98304 + (size_t)DM * DM * 2);

  mha_cvt_bf16_kernel<<<64, 256, 0, stream>>>(Wq, Wq_b, HD * HD);
  mha_cvt_bf16_kernel<<<64, 256, 0, stream>>>(Wk, Wk_b, HD * HD);
  mha_cvt_bf16_kernel<<<64, 256, 0, stream>>>(Wv, Wv_b, HD * HD);
  mha_cvt_bf16_kernel<<<4096, 256, 0, stream>>>(Wo, Wo_b, DM * DM);

  mha_fused_attn_kernel<<<NTOK / 2, 64, 2 * WAVE_LDS, stream>>>(
      q, k, v, mask, Wq_b, Wk_b, Wv_b, Cb);

  mha_out_proj_kernel<<<dim3(NTOK / 256, DM / 128), 256, 0, stream>>>(
      Cb, Wo_b, bo, (float*)d_out);
}
